// HeteroGNN_41360535060673
// MI455X (gfx1250) — compile-verified, matrix operand order checked by reading, not run on hardware
//
#include <hip/hip_runtime.h>
#include <hip/hip_bf16.h>

// HeteroGNN forward for MI455X (gfx1250, wave32, WMMA).
// All GEMMs use v_wmma_f32_16x16x32_f16 with f16 inputs / f32 accumulation.
// Workgroup = 256 threads = 8 waves; each block computes a 64x256 C tile
// (4 M-tiles x 16 N-tiles; 8 tiles per wave -> 64 accumulator VGPRs/wave).
// A/B tiles double-buffered in LDS (rows padded to 40 halves = 80B stride to
// destagger ds_load_b128 banks); one barrier per K-step. Staging copies use
// GLOBAL_LOAD_ASYNC_TO_LDS_B128 (ASYNCcnt) so staging traffic stays off DScnt
// and per-wmma waits cover only the two fragment ds_loads; drained with
// s_wait_asynccnt 0 before each step barrier.

typedef __attribute__((ext_vector_type(16))) _Float16 v16h;
typedef __attribute__((ext_vector_type(8)))  _Float16 v8h;
typedef __attribute__((ext_vector_type(8)))  float    v8f;
typedef int v4i_t __attribute__((vector_size(16)));   // builtin's pointee type

#define N_OBJ   50000
#define N_GRAPH 256
#define LDA     40            // padded row stride (halves) for 32-wide K tiles

#if defined(__has_builtin)
#if __has_builtin(__builtin_amdgcn_global_load_async_to_lds_b128)
#define HAS_ASYNC_LDS 1
#endif
#endif
#ifndef HAS_ASYNC_LDS
#define HAS_ASYNC_LDS 0
#endif

// 16B global -> LDS copy (async path on gfx1250)
__device__ __forceinline__ void copy16_g2l(const _Float16* src, _Float16* dst) {
#if HAS_ASYNC_LDS
  __builtin_amdgcn_global_load_async_to_lds_b128((v4i_t*)src, (v4i_t*)dst, 0, 0);
#else
  *(v8h*)dst = *(const v8h*)src;
#endif
}

__device__ __forceinline__ void async_wait0() {
#if HAS_ASYNC_LDS
#if __has_builtin(__builtin_amdgcn_s_wait_asynccnt)
  __builtin_amdgcn_s_wait_asynccnt(0);
#else
  asm volatile("s_wait_asynccnt 0x0" ::: "memory");
#endif
#endif
}

// Fragment load: two contiguous 8-half (16B) chunks per lane.
// CDNA5 wave32 16-bit A/B layout: lane<16 -> K{0..7,16..23},
// lane>=16 -> K{8..15,24..31} (caller offsets by half*8).
__device__ __forceinline__ v16h ld_frag(const _Float16* p) {
  v8h lo = *(const v8h*)p;
  v8h hi = *(const v8h*)(p + 16);
  v16h r;
#pragma unroll
  for (int i = 0; i < 8; ++i) { r[i] = lo[i]; r[i + 8] = hi[i]; }
  return r;
}

__device__ __forceinline__ void stage_b_row(_Float16* dst, const _Float16* src) {
#pragma unroll
  for (int j = 0; j < 4; ++j) copy16_g2l(src + j * 8, dst + j * 8);
}

// 8 wmmas for one K-step; B fragments software-pipelined 2 deep.
#define MMA_STEP(AS_, BS_)                                                     \
  do {                                                                         \
    v16h a_ = ld_frag(&(AS_)[(mt * 16 + r) * LDA + half * 8]);                 \
    v16h bcur_ = ld_frag(&(BS_)[(nb0 + r) * LDA + half * 8]);                  \
    _Pragma("unroll")                                                          \
    for (int t = 0; t < 8; ++t) {                                              \
      v16h bnext_ = bcur_;                                                     \
      if (t < 7)                                                               \
        bnext_ = ld_frag(&(BS_)[(nb0 + (t + 1) * 16 + r) * LDA + half * 8]);   \
      acc[t] = __builtin_amdgcn_wmma_f32_16x16x32_f16(                         \
          false, a_, false, bcur_, (short)0, acc[t], false, false);            \
      bcur_ = bnext_;                                                          \
    }                                                                          \
  } while (0)

// Wc[n][k] = (f16) W[k][n]   (W row-major [K][N])
__global__ __launch_bounds__(256) void k_wconv(const float* __restrict__ W,
                                               _Float16* __restrict__ Wc,
                                               int K, int N) {
  int i = blockIdx.x * 256 + threadIdx.x;
  if (i < K * N) {
    int k = i / N, n = i - k * N;
    Wc[(size_t)n * K + k] = (_Float16)W[i];
  }
}

__global__ __launch_bounds__(256) void k_zero(float* __restrict__ p, int n) {
  int i = blockIdx.x * 256 + threadIdx.x;
  if (i < n) p[i] = 0.f;
}

// -------- predicate GEMM1: H = relu(LN(gatherconcat(x, idx) @ W1 + b1)) -> f16
__global__ __launch_bounds__(256) void k_pred_g1(
    const float* __restrict__ x, const int* __restrict__ idx,
    int n_atoms, int arity,
    const _Float16* __restrict__ Wc,   // [256][K] f16, K = arity*256
    const float* __restrict__ b1, const float* __restrict__ g,
    const float* __restrict__ be, _Float16* __restrict__ Hout) {
  __shared__ __align__(16) _Float16 As[2][64 * LDA];
  __shared__ __align__(16) _Float16 Bs[2][256 * LDA];
  __shared__ int   sIdx[3 * 64];
  __shared__ float sSum[64], sSq[64];
  const int tid  = threadIdx.x;
  const int base = blockIdx.x * 64;
  const int K    = arity * 256;

  for (int t = tid; t < arity * 64; t += 256) {
    int arg = t >> 6, m = t & 63, rowg = base + m;
    sIdx[t] = (rowg < n_atoms) ? idx[(size_t)arg * n_atoms + rowg] : 0;
  }
  if (tid < 64) { sSum[tid] = 0.f; sSq[tid] = 0.f; }
  __syncthreads();

  // Warm L2 for the gathered rows (reused for all K-steps of each arg block).
  for (int t = tid; t < arity * 64 * 4; t += 256) {
    int slot = t >> 2, seg = t & 3;
    __builtin_prefetch(x + (size_t)sIdx[slot] * 256 + seg * 64, 0, 3);
  }

  const int lane = tid & 31, w = tid >> 5;
  const int half = lane >> 4, r = lane & 15;
  const int mt = w >> 1, nb0 = (w & 1) * 128;
  v8f acc[8];
#pragma unroll
  for (int t = 0; t < 8; ++t)
#pragma unroll
    for (int v = 0; v < 8; ++v) acc[t][v] = 0.f;

  const int  mstage = tid >> 2;        // 0..63
  const int  kstage = (tid & 3) * 8;   // 0,8,16,24
  const bool mvalid = (base + mstage) < n_atoms;

  auto stageA = [&](int buf, int kk) {     // gather + f32->f16 convert (VALU)
    int kg = kk + kstage, arg = kg >> 8, koff = kg & 255;
    const float* src = x + (size_t)sIdx[arg * 64 + mstage] * 256 + koff;
    _Float16* dst = &As[buf][mstage * LDA + kstage];
#pragma unroll
    for (int j = 0; j < 8; ++j) dst[j] = mvalid ? (_Float16)src[j] : (_Float16)0.f;
  };
  auto stageB = [&](int buf, int kk) {
    stage_b_row(&Bs[buf][tid * LDA], Wc + (size_t)tid * K + kk);
  };

  stageA(0, 0); stageB(0, 0);
  const int nsteps = K >> 5;
  for (int s = 0; s < nsteps; ++s) {
    async_wait0();
    __syncthreads();
    int buf = s & 1;
    if (s + 1 < nsteps) { stageA(buf ^ 1, (s + 1) * 32); stageB(buf ^ 1, (s + 1) * 32); }
    MMA_STEP(As[buf], Bs[buf]);
  }
  // bias, then per-row LayerNorm stats via LDS float atomics
#pragma unroll
  for (int t = 0; t < 8; ++t) {
    float bv = b1[nb0 + t * 16 + r];
#pragma unroll
    for (int v = 0; v < 8; ++v) acc[t][v] += bv;
  }
#pragma unroll
  for (int v = 0; v < 8; ++v) {
    float s = 0.f, sq = 0.f;
#pragma unroll
    for (int t = 0; t < 8; ++t) { float val = acc[t][v]; s += val; sq += val * val; }
    int m = mt * 16 + half * 8 + v;
    atomicAdd(&sSum[m], s);
    atomicAdd(&sSq[m], sq);
  }
  __syncthreads();
#pragma unroll
  for (int v = 0; v < 8; ++v) {
    int m = mt * 16 + half * 8 + v;
    int rowg = base + m;
    float mu   = sSum[m] * (1.f / 256.f);
    float var  = sSq[m] * (1.f / 256.f) - mu * mu;
    float rstd = rsqrtf(var + 1e-5f);
    if (rowg < n_atoms) {
#pragma unroll
      for (int t = 0; t < 8; ++t) {
        int n = nb0 + t * 16 + r;
        float val = (acc[t][v] - mu) * rstd * g[n] + be[n];
        val = fmaxf(val, 0.f);
        Hout[(size_t)rowg * 256 + n] = (_Float16)val;
      }
    }
  }
}

// -------- predicate GEMM2 + scatter: msg[idx[arg][m]] += (H @ W2 + b2)[:, arg blk]
__global__ __launch_bounds__(256) void k_pred_g2(
    const _Float16* __restrict__ H, const int* __restrict__ idx, int n_atoms,
    const _Float16* __restrict__ Wc,   // [arity*256][256] f16
    const float* __restrict__ b2, float* __restrict__ msg) {
  __shared__ __align__(16) _Float16 As[2][64 * LDA];
  __shared__ __align__(16) _Float16 Bs[2][256 * LDA];
  __shared__ int sIdx[64];
  const int tid  = threadIdx.x;
  const int base = blockIdx.x * 64;
  const int arg  = blockIdx.y;
  if (tid < 64)
    sIdx[tid] = (base + tid < n_atoms) ? idx[(size_t)arg * n_atoms + base + tid] : 0;

  const int lane = tid & 31, w = tid >> 5;
  const int half = lane >> 4, r = lane & 15;
  const int mt = w >> 1, nb0 = (w & 1) * 128;
  v8f acc[8];
#pragma unroll
  for (int t = 0; t < 8; ++t)
#pragma unroll
    for (int v = 0; v < 8; ++v) acc[t][v] = 0.f;

  const int  mstage = tid >> 2;
  const int  kstage = (tid & 3) * 8;
  const bool mvalid = (base + mstage) < n_atoms;

  auto stageA = [&](int buf, int kk) {
    _Float16* dst = &As[buf][mstage * LDA + kstage];
    if (mvalid) {
      copy16_g2l(H + (size_t)(base + mstage) * 256 + kk + kstage, dst);
    } else {
      v8h z;
#pragma unroll
      for (int j = 0; j < 8; ++j) z[j] = (_Float16)0.f;
      *(v8h*)dst = z;
    }
  };
  auto stageB = [&](int buf, int kk) {
    stage_b_row(&Bs[buf][tid * LDA],
                Wc + (size_t)(arg * 256 + tid) * 256 + kk);
  };

  stageA(0, 0); stageB(0, 0);
  for (int s = 0; s < 8; ++s) {
    async_wait0();
    __syncthreads();
    int buf = s & 1;
    if (s + 1 < 8) { stageA(buf ^ 1, (s + 1) * 32); stageB(buf ^ 1, (s + 1) * 32); }
    MMA_STEP(As[buf], Bs[buf]);
  }
#pragma unroll
  for (int v = 0; v < 8; ++v) {
    int m = mt * 16 + half * 8 + v;
    int rowg = base + m;
    if (rowg < n_atoms) {
      int trow = sIdx[m];
#pragma unroll
      for (int t = 0; t < 8; ++t) {
        int nloc = nb0 + t * 16 + r;
        float val = acc[t][v] + b2[arg * 256 + nloc];
        atomicAdd(&msg[(size_t)trow * 256 + nloc], val);
      }
    }
  }
}

// -------- obj update GEMM1: Hpre = concat(x,msg) @ upW1 + b1 ; column stats
__global__ __launch_bounds__(256) void k_up1(
    const float* __restrict__ x, const float* __restrict__ msg,
    const _Float16* __restrict__ Wc,   // [512][512] f16
    const float* __restrict__ b1, float* __restrict__ Hpre,
    float* __restrict__ gstats) {
  __shared__ __align__(16) _Float16 As[2][64 * LDA];
  __shared__ __align__(16) _Float16 Bs[2][256 * LDA];
  const int tid    = threadIdx.x;
  const int base   = blockIdx.x * 64;
  const int nchunk = blockIdx.y;   // 0 or 1

  const int lane = tid & 31, w = tid >> 5;
  const int half = lane >> 4, r = lane & 15;
  const int mt = w >> 1, nb0 = (w & 1) * 128;
  v8f acc[8];
#pragma unroll
  for (int t = 0; t < 8; ++t)
#pragma unroll
    for (int v = 0; v < 8; ++v) acc[t][v] = 0.f;

  const int  mstage = tid >> 2;
  const int  kstage = (tid & 3) * 8;
  const bool mvalid = (base + mstage) < N_OBJ;

  auto stageA = [&](int buf, int kk) {     // concat + f32->f16 convert (VALU)
    int kg = kk + kstage;
    const float* src = (kg < 256) ? (x + (size_t)(base + mstage) * 256 + kg)
                                  : (msg + (size_t)(base + mstage) * 256 + (kg - 256));
    _Float16* dst = &As[buf][mstage * LDA + kstage];
#pragma unroll
    for (int j = 0; j < 8; ++j) dst[j] = mvalid ? (_Float16)src[j] : (_Float16)0.f;
  };
  auto stageB = [&](int buf, int kk) {
    stage_b_row(&Bs[buf][tid * LDA],
                Wc + (size_t)(nchunk * 256 + tid) * 512 + kk);
  };

  stageA(0, 0); stageB(0, 0);
  for (int s = 0; s < 16; ++s) {
    async_wait0();
    __syncthreads();
    int buf = s & 1;
    if (s + 1 < 16) { stageA(buf ^ 1, (s + 1) * 32); stageB(buf ^ 1, (s + 1) * 32); }
    MMA_STEP(As[buf], Bs[buf]);
  }
#pragma unroll
  for (int t = 0; t < 8; ++t) {
    int n = nchunk * 256 + nb0 + t * 16 + r;
    float bv = b1[n];
    float s = 0.f, sq = 0.f;
#pragma unroll
    for (int v = 0; v < 8; ++v) {
      int rowg = base + mt * 16 + half * 8 + v;
      float val = acc[t][v] + bv;
      if (rowg < N_OBJ) {
        Hpre[(size_t)rowg * 512 + n] = val;
        s += val; sq += val * val;
      }
    }
    atomicAdd(&gstats[n], s);
    atomicAdd(&gstats[512 + n], sq);
  }
}

__global__ __launch_bounds__(512) void k_bn_finalize(const float* __restrict__ gstats,
                                                     float* __restrict__ gnorm) {
  int n = threadIdx.x;
  float mu  = gstats[n] * (1.f / (float)N_OBJ);
  float var = gstats[512 + n] * (1.f / (float)N_OBJ) - mu * mu;
  gnorm[n]       = mu;
  gnorm[512 + n] = rsqrtf(var + 1e-5f);
}

__global__ __launch_bounds__(256) void k_bn_relu(
    const float* __restrict__ Hpre, const float* __restrict__ gnorm,
    const float* __restrict__ g, const float* __restrict__ be,
    _Float16* __restrict__ Hf16) {
  int i = blockIdx.x * 256 + threadIdx.x;   // < N_OBJ*512 (exact)
  int n = i & 511;
  float v = (Hpre[i] - gnorm[n]) * gnorm[512 + n] * g[n] + be[n];
  Hf16[i] = (_Float16)fmaxf(v, 0.f);
}

// -------- obj update GEMM2: x_out = Hf16 @ upW2 + b2
__global__ __launch_bounds__(256) void k_up2(
    const _Float16* __restrict__ H, const _Float16* __restrict__ Wc,  // [256][512]
    const float* __restrict__ b2, float* __restrict__ xout) {
  __shared__ __align__(16) _Float16 As[2][64 * LDA];
  __shared__ __align__(16) _Float16 Bs[2][256 * LDA];
  const int tid  = threadIdx.x;
  const int base = blockIdx.x * 64;
  const int lane = tid & 31, w = tid >> 5;
  const int half = lane >> 4, r = lane & 15;
  const int mt = w >> 1, nb0 = (w & 1) * 128;
  v8f acc[8];
#pragma unroll
  for (int t = 0; t < 8; ++t)
#pragma unroll
    for (int v = 0; v < 8; ++v) acc[t][v] = 0.f;

  const int  mstage = tid >> 2;
  const int  kstage = (tid & 3) * 8;
  const bool mvalid = (base + mstage) < N_OBJ;

  auto stageA = [&](int buf, int kk) {
    _Float16* dst = &As[buf][mstage * LDA + kstage];
    if (mvalid) {
      copy16_g2l(H + (size_t)(base + mstage) * 512 + kk + kstage, dst);
    } else {
      v8h z;
#pragma unroll
      for (int j = 0; j < 8; ++j) z[j] = (_Float16)0.f;
      *(v8h*)dst = z;
    }
  };
  auto stageB = [&](int buf, int kk) {
    stage_b_row(&Bs[buf][tid * LDA], Wc + (size_t)tid * 512 + kk);
  };

  stageA(0, 0); stageB(0, 0);
  for (int s = 0; s < 16; ++s) {
    async_wait0();
    __syncthreads();
    int buf = s & 1;
    if (s + 1 < 16) { stageA(buf ^ 1, (s + 1) * 32); stageB(buf ^ 1, (s + 1) * 32); }
    MMA_STEP(As[buf], Bs[buf]);
  }
#pragma unroll
  for (int v = 0; v < 8; ++v) {
    int rowg = base + mt * 16 + half * 8 + v;
    if (rowg < N_OBJ) {
#pragma unroll
      for (int t = 0; t < 8; ++t) {
        int n = nb0 + t * 16 + r;
        xout[(size_t)rowg * 256 + n] = acc[t][v] + b2[n];
      }
    }
  }
}

// -------- global add pool by batch id
__global__ __launch_bounds__(256) void k_pool(const float* __restrict__ x,
                                              const int* __restrict__ batch,
                                              float* __restrict__ aggr) {
  int i = blockIdx.x * 256 + threadIdx.x;   // < N_OBJ*256 (exact)
  int o = i >> 8, c = i & 255;
  atomicAdd(&aggr[(size_t)batch[o] * 256 + c], x[i]);
}

// -------- readout MLP: per-graph  mish(LN(a@W1+b1)) @ W2 + b2
__global__ __launch_bounds__(512) void k_readout(
    const float* __restrict__ aggr,
    const float* __restrict__ W1, const float* __restrict__ b1,
    const float* __restrict__ g, const float* __restrict__ be,
    const float* __restrict__ W2, const float* __restrict__ b2,
    float* __restrict__ out) {
  __shared__ float sA[256];
  __shared__ float sRed[3];
  const int tid = threadIdx.x;
  const int gr  = blockIdx.x;
  if (tid < 256) sA[tid] = aggr[(size_t)gr * 256 + tid];
  if (tid < 3) sRed[tid] = 0.f;
  __syncthreads();
  float h = b1[tid];
  for (int k = 0; k < 256; ++k) h += sA[k] * W1[(size_t)k * 512 + tid];
  atomicAdd(&sRed[0], h);
  atomicAdd(&sRed[1], h * h);
  __syncthreads();
  float mu   = sRed[0] * (1.f / 512.f);
  float var  = sRed[1] * (1.f / 512.f) - mu * mu;
  float rstd = rsqrtf(var + 1e-5f);
  float hn = (h - mu) * rstd * g[tid] + be[tid];
  float sp = (hn > 20.f) ? hn : log1pf(expf(hn));
  float mish = hn * tanhf(sp);
  atomicAdd(&sRed[2], mish * W2[tid]);
  __syncthreads();
  if (tid == 0) out[gr] = sRed[2] + b2[0];
}

// =================== host launcher ===================
extern "C" void kernel_launch(void* const* d_in, const int* in_sizes, int n_in,
                              void* d_out, int out_size, void* d_ws, size_t ws_size,
                              hipStream_t stream) {
  (void)in_sizes; (void)n_in; (void)out_size; (void)ws_size;
  const float* x0    = (const float*)d_in[0];
  const int*   ip1   = (const int*)d_in[1];
  const int*   ip2   = (const int*)d_in[2];
  const int*   ip3   = (const int*)d_in[3];
  const int*   batch = (const int*)d_in[4];
  // 6 params per group: W1,b1,g,be,W2,b2
  const float* P[5][6];
  for (int grp = 0; grp < 5; ++grp)
    for (int j = 0; j < 6; ++j)
      P[grp][j] = (const float*)d_in[5 + grp * 6 + j];
  // grp: 0=p1, 1=p2, 2=p3, 3=up, 4=ro

  char* ws = (char*)d_ws;
  float*    xbuf   = (float*)(ws + 0);                       // 50000*256 f32
  float*    msg    = (float*)(ws + 51200000);                // 50000*256 f32
  float*    Hpre   = (float*)(ws + 102400000);               // 50000*512 f32
  _Float16* Hf16   = (_Float16*)(ws + 204800000);            // 50000*512 f16
  float*    gstats = (float*)(ws + 256000000);               // 1024 f32
  float*    gnorm  = (float*)(ws + 256004096);               // 1024 f32
  float*    aggr   = (float*)(ws + 256008192);               // 256*256 f32
  _Float16* wcb    = (_Float16*)(ws + 256270336);            // f16 weights
  _Float16* p1W1c = wcb + 0;
  _Float16* p2W1c = wcb + 65536;
  _Float16* p3W1c = wcb + 196608;
  _Float16* p1W2c = wcb + 393216;
  _Float16* p2W2c = wcb + 458752;
  _Float16* p3W2c = wcb + 589824;
  _Float16* upW1c = wcb + 786432;
  _Float16* upW2c = wcb + 1048576;

  // weight transpose + f16 convert (deterministic every call)
  {
    struct { const float* W; _Float16* Wc; int K, N; } cv[8] = {
      { P[0][0], p1W1c, 256, 256 }, { P[1][0], p2W1c, 512, 256 },
      { P[2][0], p3W1c, 768, 256 }, { P[0][4], p1W2c, 256, 256 },
      { P[1][4], p2W2c, 256, 512 }, { P[2][4], p3W2c, 256, 768 },
      { P[3][0], upW1c, 512, 512 }, { P[3][4], upW2c, 512, 256 },
    };
    for (int i = 0; i < 8; ++i)
      k_wconv<<<(cv[i].K * cv[i].N + 255) / 256, 256, 0, stream>>>(
          cv[i].W, cv[i].Wc, cv[i].K, cv[i].N);
  }

  const int   na[3]    = { 40000, 60000, 30000 };
  const int*  idxs[3]  = { ip1, ip2, ip3 };
  _Float16*   w1c[3]   = { p1W1c, p2W1c, p3W1c };
  _Float16*   w2c[3]   = { p1W2c, p2W2c, p3W2c };

  const float* xcur = x0;
  for (int layer = 0; layer < 3; ++layer) {
    k_zero<<<(N_OBJ * 256 + 255) / 256, 256, 0, stream>>>(msg, N_OBJ * 256);
    k_zero<<<4, 256, 0, stream>>>(gstats, 1024);
    for (int p = 0; p < 3; ++p) {
      int arity = p + 1;
      int gx = (na[p] + 63) / 64;
      k_pred_g1<<<gx, 256, 0, stream>>>(xcur, idxs[p], na[p], arity, w1c[p],
                                        P[p][1], P[p][2], P[p][3], Hf16);
      k_pred_g2<<<dim3(gx, arity), 256, 0, stream>>>(Hf16, idxs[p], na[p],
                                                     w2c[p], P[p][5], msg);
    }
    int gu = (N_OBJ + 63) / 64;
    k_up1<<<dim3(gu, 2), 256, 0, stream>>>(xcur, msg, upW1c, P[3][1], Hpre, gstats);
    k_bn_finalize<<<1, 512, 0, stream>>>(gstats, gnorm);
    k_bn_relu<<<(N_OBJ * 512) / 256, 256, 0, stream>>>(Hpre, gnorm, P[3][2],
                                                       P[3][3], Hf16);
    k_up2<<<gu, 256, 0, stream>>>(Hf16, upW2c, P[3][5], xbuf);
    xcur = xbuf;
  }

  k_zero<<<(N_GRAPH * 256 + 255) / 256, 256, 0, stream>>>(aggr, N_GRAPH * 256);
  k_pool<<<(N_OBJ * 256) / 256, 256, 0, stream>>>(xcur, batch, aggr);
  k_readout<<<N_GRAPH, 512, 0, stream>>>(aggr, P[4][0], P[4][1], P[4][2],
                                         P[4][3], P[4][4], P[4][5],
                                         (float*)d_out);
}
